// MossVLSelfAttention_36936718746211
// MI455X (gfx1250) — matrix operand order repaired and checked by reading, not compiled, not run on hardware
//
#include <hip/hip_runtime.h>
#include <math.h>

// ---------------------------------------------------------------------------
// MossVLSelfAttention for MI455X (gfx1250), wave32 + WMMA f32_16x16x32_f16.
// ---------------------------------------------------------------------------

typedef __attribute__((ext_vector_type(16))) _Float16 v16h;
typedef __attribute__((ext_vector_type(8)))  _Float16 v8h;
typedef __attribute__((ext_vector_type(4)))  _Float16 v4h;
typedef __attribute__((ext_vector_type(8)))  float    v8f;

#define NHD    16     // query heads
#define NKVH   8      // kv heads
#define DHEAD  128    // head dim
#define HIDDEN 2048
#define QKVN   ((NHD + 2 * NKVH) * DHEAD)   // 4096

static __device__ __forceinline__ v16h make_frag(const _Float16* lo_p,
                                                 const _Float16* hi_p) {
  v8h lo = *reinterpret_cast<const v8h*>(lo_p);
  v8h hi = *reinterpret_cast<const v8h*>(hi_p);
  v16h r;
#pragma unroll
  for (int e = 0; e < 8; ++e) { r[e] = lo[e]; r[e + 8] = hi[e]; }
  return r;
}

// ---------------------------------------------------------------------------
// fp32-in / fp32-out GEMM, f16 WMMA, f32 accumulate.
// Block tile 128x128, K-step 32, 8 waves, wave tile 32x64.
// Double-buffered LDS: global loads for tile(i+1) issue before the WMMAs of
// tile(i); global_prefetch for tile(i+2).
// ---------------------------------------------------------------------------
__launch_bounds__(256)
__global__ void wmma_gemm_f32(const float* __restrict__ A,
                              const float* __restrict__ B,
                              float* __restrict__ C,
                              int M, int N, int K) {
  __shared__ __align__(16) _Float16 As[2][128][32];   // [buf][m][k]
  __shared__ __align__(16) _Float16 Bs[2][128][32];   // [buf][n][k] (transposed)

  const int tid  = threadIdx.x;
  const int lane = tid & 31;
  const int wid  = tid >> 5;
  const int hs   = lane >> 4;      // half-wave select
  const int l15  = lane & 15;

  const int bm = blockIdx.y * 128;
  const int bn = blockIdx.x * 128;
  const int wm = (wid & 3) * 32;   // wave row offset inside tile
  const int wn = (wid >> 2) * 64;  // wave col offset inside tile

  // Per-thread staging coordinates: each thread owns 64 contiguous bytes.
  const int ra  = (tid * 4) >> 3;          // A row (0..127)
  const int ca  = ((tid * 4) & 7) * 4;     // A col base (0 or 16)
  const int kkb = (tid * 4) >> 5;          // B k-row (0..31)
  const int nb  = ((tid * 4) & 31) * 4;    // B n base (0..112)

  v8f acc[2][4];
#pragma unroll
  for (int ti = 0; ti < 2; ++ti)
#pragma unroll
    for (int tj = 0; tj < 4; ++tj)
#pragma unroll
      for (int e = 0; e < 8; ++e) acc[ti][tj][e] = 0.0f;

  float4 ar[4], br[4];
  const float* Arow = A + (size_t)(bm + ra) * K + ca;
  const float* Bcol = B + bn + nb;

  auto load_tiles = [&](int k0) {
#pragma unroll
    for (int l = 0; l < 4; ++l)
      ar[l] = *reinterpret_cast<const float4*>(Arow + k0 + 4 * l);
#pragma unroll
    for (int l = 0; l < 4; ++l)
      br[l] = *reinterpret_cast<const float4*>(Bcol + (size_t)(k0 + kkb) * N + 4 * l);
  };
  auto store_tiles = [&](int buf) {
#pragma unroll
    for (int l = 0; l < 4; ++l) {
      As[buf][ra][ca + 4 * l + 0] = (_Float16)ar[l].x;
      As[buf][ra][ca + 4 * l + 1] = (_Float16)ar[l].y;
      As[buf][ra][ca + 4 * l + 2] = (_Float16)ar[l].z;
      As[buf][ra][ca + 4 * l + 3] = (_Float16)ar[l].w;
    }
#pragma unroll
    for (int l = 0; l < 4; ++l) {
      Bs[buf][nb + 4 * l + 0][kkb] = (_Float16)br[l].x;
      Bs[buf][nb + 4 * l + 1][kkb] = (_Float16)br[l].y;
      Bs[buf][nb + 4 * l + 2][kkb] = (_Float16)br[l].z;
      Bs[buf][nb + 4 * l + 3][kkb] = (_Float16)br[l].w;
    }
  };

  const int NIT = K >> 5;
  load_tiles(0);
  store_tiles(0);
  __syncthreads();

  for (int it = 0; it < NIT; ++it) {
    const int cur = it & 1;
    if (it + 1 < NIT) load_tiles((it + 1) * 32);       // overlap with WMMAs
    if (it + 2 < NIT) {                                // global_prefetch_b8
      __builtin_prefetch(Arow + (it + 2) * 32, 0, 3);
      __builtin_prefetch(Bcol + (size_t)((it + 2) * 32 + kkb) * N, 0, 3);
    }

    // A fragments (16x32, 16-bit): lanes<16 hold K 0..7 / 16..23, lanes>=16
    // hold K 8..15 / 24..31 (ISA 7.12.2).
    v16h af[2];
#pragma unroll
    for (int ti = 0; ti < 2; ++ti) {
      const _Float16* p = &As[cur][wm + ti * 16 + l15][0];
      af[ti] = make_frag(p + hs * 8, p + 16 + hs * 8);
    }
    // B fragments (32x16): lanes<16 hold K 0..7 / 16..23, lanes>=16 hold K 8..15 / 24..31.
    v16h bf[4];
#pragma unroll
    for (int tj = 0; tj < 4; ++tj) {
      const _Float16* p = &Bs[cur][wn + tj * 16 + l15][0];
      bf[tj] = make_frag(p + hs * 8, p + 16 + hs * 8);
    }
#pragma unroll
    for (int ti = 0; ti < 2; ++ti)
#pragma unroll
      for (int tj = 0; tj < 4; ++tj)
        acc[ti][tj] = __builtin_amdgcn_wmma_f32_16x16x32_f16(
            false, af[ti], false, bf[tj], (short)0, acc[ti][tj], false, false);

    __syncthreads();
    if (it + 1 < NIT) store_tiles(cur ^ 1);
    __syncthreads();
  }

  // Write out: C/D layout — VGPR e, lanes<16 -> M=e, lanes>=16 -> M=8+e.
#pragma unroll
  for (int ti = 0; ti < 2; ++ti)
#pragma unroll
    for (int tj = 0; tj < 4; ++tj) {
      int r0 = bm + wm + ti * 16 + hs * 8;
      int c  = bn + wn + tj * 16 + l15;
#pragma unroll
      for (int e = 0; e < 8; ++e)
        C[(size_t)(r0 + e) * N + c] = acc[ti][tj][e];
    }
}

// ---------------------------------------------------------------------------
// RMSNorm + RoPE + precision/layout packing.
// One wave per (token, slice); slice: 0..15 -> Q head, 16..23 -> K head,
// 24..31 -> V head. Lane owns 4 contiguous dims.
// Q -> [NHD][T][D] f16, K -> [NKVH][T][D] f16, V -> [NKVH][D][T] f16
// (transposed so PV B-fragments are contiguous in the key dimension).
// ---------------------------------------------------------------------------
__launch_bounds__(256)
__global__ void norm_rope_pack(const float* __restrict__ qkv,
                               const float* __restrict__ qw,
                               const float* __restrict__ kw,
                               const int* __restrict__ positions,
                               _Float16* __restrict__ Qh,
                               _Float16* __restrict__ Kh,
                               _Float16* __restrict__ Vt,
                               int T) {
  const int gw    = blockIdx.x * 8 + (threadIdx.x >> 5);
  const int lane  = threadIdx.x & 31;
  const int t     = gw >> 5;
  const int slice = gw & 31;
  if (t >= T) return;

  const float* src = qkv + (size_t)t * QKVN + slice * DHEAD;
  const int d0 = lane * 4;
  float4 xv = *reinterpret_cast<const float4*>(src + d0);
  float x[4] = {xv.x, xv.y, xv.z, xv.w};

  if (slice >= 24) {  // V: no norm, no rope, store transposed
    int h = slice - 24;
#pragma unroll
    for (int j = 0; j < 4; ++j)
      Vt[(size_t)(h * DHEAD + d0 + j) * T + t] = (_Float16)x[j];
    return;
  }

  // RMSNorm over 128 dims (wave reduction)
  float ss = x[0] * x[0] + x[1] * x[1] + x[2] * x[2] + x[3] * x[3];
#pragma unroll
  for (int m = 1; m < 32; m <<= 1) ss += __shfl_xor(ss, m, 32);
  const float rms = rsqrtf(ss * (1.0f / 128.0f) + 1e-6f);
  const float* w = (slice < 16) ? qw : kw;

  float xn[4];
#pragma unroll
  for (int j = 0; j < 4; ++j) xn[j] = x[j] * rms * w[d0 + j];

  // RoPE: halves are 64 apart => partner lane is lane ^ 16.
  const float pos = (float)positions[t];
  const float kfreq = 0.21586206f;   // ln(1e6)/64
  float out[4];
#pragma unroll
  for (int j = 0; j < 4; ++j) {
    float p  = __shfl_xor(xn[j], 16, 32);
    int   dd = (d0 + j) & 63;
    float ang = pos * __expf(-(float)dd * kfreq);
    float c, s;
    __sincosf(ang, &s, &c);
    // lane<16: x1*c - x2*s ; lane>=16: x2*c + x1*s
    out[j] = (lane < 16) ? (xn[j] * c - p * s) : (xn[j] * c + p * s);
  }

  _Float16* dst = (slice < 16)
      ? Qh + ((size_t)slice * T + t) * DHEAD + d0
      : Kh + ((size_t)(slice - 16) * T + t) * DHEAD + d0;
#pragma unroll
  for (int j = 0; j < 4; ++j) dst[j] = (_Float16)out[j];
}

// ---------------------------------------------------------------------------
// Causal flash attention, GQA 2:1. Block = 4 waves x 16 query rows = 64 rows.
// Per-wave causal loop bound => per-wave LDS scratch, no block barriers in
// the key loop (s_wait_dscnt keeps per-wave LDS RAW ordered).
// ---------------------------------------------------------------------------
__launch_bounds__(128)
__global__ void flash_attn(const _Float16* __restrict__ Qh,
                           const _Float16* __restrict__ Kh,
                           const _Float16* __restrict__ Vt,
                           float* __restrict__ attn,
                           int T, float scale) {
  __shared__ __align__(16) float    Sbuf[4][16][32];
  __shared__ __align__(16) _Float16 Pbuf[4][16][32];
  __shared__ float lm[4][16], ll[4][16], lsc[4][16];

  const int lane = threadIdx.x & 31;
  const int wid  = threadIdx.x >> 5;
  const int hs   = lane >> 4;
  const int l15  = lane & 15;
  const int h    = blockIdx.y;
  const int kvh  = h >> 1;                 // NH/NKV == 2
  const int q0   = blockIdx.x * 64 + wid * 16;
  const int qhi  = q0 + 15;

  // Preload Q fragments: Q[16 x 128] as 4 A-fragments (K-chunks of 32).
  v16h qf[4];
  {
    const _Float16* qp = Qh + ((size_t)h * T + q0 + l15) * DHEAD;
#pragma unroll
    for (int dk = 0; dk < 4; ++dk)
      qf[dk] = make_frag(qp + dk * 32 + hs * 8, qp + dk * 32 + 16 + hs * 8);
  }

  v8f o[8];
#pragma unroll
  for (int dn = 0; dn < 8; ++dn)
#pragma unroll
    for (int e = 0; e < 8; ++e) o[dn][e] = 0.0f;

  if (lane < 16) { lm[wid][lane] = -1e30f; ll[wid][lane] = 0.0f; }
  asm volatile("s_wait_dscnt 0" ::: "memory");

  for (int kb = 0; kb <= qhi; kb += 32) {
    // Prefetch next key block of K and (sparsely) V.
    if (kb + 32 <= qhi) {
      __builtin_prefetch(Kh + ((size_t)kvh * T + kb + 32 + l15) * DHEAD, 0, 3);
      __builtin_prefetch(Vt + (size_t)(kvh * DHEAD + lane * 4) * T + kb + 32, 0, 3);
    }

    // ---- S = Q K^T for two 16-key sub-tiles --------------------------------
#pragma unroll
    for (int t2 = 0; t2 < 2; ++t2) {
      const int kbase = kb + t2 * 16;
      v8f s;
#pragma unroll
      for (int e = 0; e < 8; ++e) s[e] = 0.0f;
      const _Float16* kp = Kh + ((size_t)kvh * T + kbase + l15) * DHEAD;
#pragma unroll
      for (int dk = 0; dk < 4; ++dk) {
        v16h bfr = make_frag(kp + dk * 32 + hs * 8, kp + dk * 32 + 16 + hs * 8);
        s = __builtin_amdgcn_wmma_f32_16x16x32_f16(
            false, qf[dk], false, bfr, (short)0, s, false, false);
      }
      const int key = kbase + l15;
#pragma unroll
      for (int e = 0; e < 8; ++e) {
        int row = hs * 8 + e;
        float v = (key <= q0 + row) ? s[e] * scale : -1e30f;
        Sbuf[wid][row][t2 * 16 + l15] = v;
      }
    }
    asm volatile("s_wait_dscnt 0" ::: "memory");

    // ---- Online softmax (lanes 0..15 own one row each) ---------------------
    if (lane < 16) {
      const int row = lane;
      float4 sv[8];
      const float4* srow = reinterpret_cast<const float4*>(&Sbuf[wid][row][0]);
#pragma unroll
      for (int j = 0; j < 8; ++j) sv[j] = srow[j];
      float mo = lm[wid][row];
      float mx = mo;
#pragma unroll
      for (int j = 0; j < 8; ++j)
        mx = fmaxf(mx, fmaxf(fmaxf(sv[j].x, sv[j].y), fmaxf(sv[j].z, sv[j].w)));
      float corr = __expf(mo - mx);
      float lsum = ll[wid][row] * corr;
      _Float16* prow = &Pbuf[wid][row][0];
#pragma unroll
      for (int j = 0; j < 8; ++j) {
        float p0 = __expf(sv[j].x - mx);
        float p1 = __expf(sv[j].y - mx);
        float p2 = __expf(sv[j].z - mx);
        float p3 = __expf(sv[j].w - mx);
        lsum += (p0 + p1) + (p2 + p3);
        v4h pk = {(_Float16)p0, (_Float16)p1, (_Float16)p2, (_Float16)p3};
        *reinterpret_cast<v4h*>(prow + 4 * j) = pk;   // one ds_store_b64
      }
      lm[wid][row]  = mx;
      ll[wid][row]  = lsum;
      lsc[wid][row] = corr;
    }
    asm volatile("s_wait_dscnt 0" ::: "memory");

    // ---- O = diag(corr)*O + P V -------------------------------------------
    float cr[8];
#pragma unroll
    for (int e = 0; e < 8; ++e) cr[e] = lsc[wid][hs * 8 + e];

    const _Float16* pp = &Pbuf[wid][l15][0];
    v16h pf = make_frag(pp + hs * 8, pp + 16 + hs * 8);

#pragma unroll
    for (int dn = 0; dn < 8; ++dn) {
      const _Float16* vp =
          Vt + (size_t)(kvh * DHEAD + dn * 16 + l15) * T + kb;
      v16h vf = make_frag(vp + hs * 8, vp + 16 + hs * 8);
#pragma unroll
      for (int e = 0; e < 8; ++e) o[dn][e] *= cr[e];
      o[dn] = __builtin_amdgcn_wmma_f32_16x16x32_f16(
          false, pf, false, vf, (short)0, o[dn], false, false);
    }
  }

  // ---- Finalize: divide by l, write attn [T][NH*D] fp32 --------------------
  float il[8];
#pragma unroll
  for (int e = 0; e < 8; ++e) il[e] = 1.0f / ll[wid][hs * 8 + e];
#pragma unroll
  for (int dn = 0; dn < 8; ++dn)
#pragma unroll
    for (int e = 0; e < 8; ++e)
      attn[(size_t)(q0 + hs * 8 + e) * (NHD * DHEAD) + h * DHEAD + dn * 16 + l15] =
          o[dn][e] * il[e];
}

// ---------------------------------------------------------------------------
// Host launcher
// ---------------------------------------------------------------------------
extern "C" void kernel_launch(void* const* d_in, const int* in_sizes, int n_in,
                              void* d_out, int out_size, void* d_ws, size_t ws_size,
                              hipStream_t stream) {
  const int*   positions = (const int*)d_in[0];
  const float* hidden    = (const float*)d_in[1];
  const float* w_qkv     = (const float*)d_in[2];
  const float* qw        = (const float*)d_in[3];
  const float* kw        = (const float*)d_in[4];
  const float* w_o       = (const float*)d_in[5];
  float*       out       = (float*)d_out;

  const int T = in_sizes[0];   // 2048

  char* ws = (char*)d_ws;
  size_t off = 0;
  float* qkv = (float*)(ws + off);       off += (size_t)T * QKVN * sizeof(float);
  _Float16* Qh = (_Float16*)(ws + off);  off += (size_t)NHD  * T * DHEAD * sizeof(_Float16);
  _Float16* Kh = (_Float16*)(ws + off);  off += (size_t)NKVH * T * DHEAD * sizeof(_Float16);
  _Float16* Vt = (_Float16*)(ws + off);  off += (size_t)NKVH * DHEAD * T * sizeof(_Float16);
  float* attn = (float*)(ws + off);      off += (size_t)T * NHD * DHEAD * sizeof(float);

  // 1) QKV projection: [T,HID] x [HID,QKVN]
  wmma_gemm_f32<<<dim3(QKVN / 128, T / 128), 256, 0, stream>>>(
      hidden, w_qkv, qkv, T, QKVN, HIDDEN);

  // 2) RMSNorm + RoPE + pack f16 (V transposed)
  norm_rope_pack<<<dim3(T * 4), 256, 0, stream>>>(
      qkv, qw, kw, positions, Qh, Kh, Vt, T);

  // 3) Causal flash attention
  flash_attn<<<dim3(T / 64, NHD), 128, 0, stream>>>(
      Qh, Kh, Vt, attn, T, 1.0f / sqrtf((float)DHEAD));

  // 4) Output projection: [T, NH*D] x [NH*D, HID]
  wmma_gemm_f32<<<dim3(HIDDEN / 128, T / 128), 256, 0, stream>>>(
      attn, w_o, out, T, HIDDEN, NHD * DHEAD);
}
